// MllamaTextCrossAttention_89696097009811
// MI455X (gfx1250) — matrix operand order repaired and checked by reading, not compiled, not run on hardware
//
#include <hip/hip_runtime.h>
#include <hip/hip_bf16.h>
#include <stdint.h>

typedef __attribute__((ext_vector_type(16))) _Float16 v16h;
typedef __attribute__((ext_vector_type(8)))  _Float16 v8h;
typedef __attribute__((ext_vector_type(4)))  _Float16 v4h;
typedef __attribute__((ext_vector_type(8)))  float    v8f;
typedef __attribute__((ext_vector_type(4)))  int      v4i;

#define H_     32
#define KVH_   8
#define G_     4
#define D_     128
#define HID_   4096
#define HD_    4096   /* H*D  */
#define KVD_   1024   /* KVH*D */
#define QKVN_  6144   /* (H+2*KVH)*D */
#define EPS_   1e-5f
#define SCALE_ 0.08838834764831845f /* 128^-0.5 */

// ---------------- CDNA5 async global->LDS copy (ASYNCcnt) ----------------
#if defined(__has_builtin)
#if __has_builtin(__builtin_amdgcn_global_load_async_to_lds_b128)
#define HAVE_ASYNC_BUILTIN 1
#endif
#if __has_builtin(__builtin_amdgcn_s_wait_asynccnt)
#define HAVE_WAITASYNC_BUILTIN 1
#endif
#if __has_builtin(__builtin_amdgcn_permlane16)
#define HAVE_PERMLANE16 1
#endif
#endif

typedef __attribute__((address_space(1))) v4i as1_v4i;   // global 128-bit unit
typedef __attribute__((address_space(3))) v4i as3_v4i;   // LDS    128-bit unit

__device__ __forceinline__ void async_cp_b128(const void* g, void* l) {
#ifdef HAVE_ASYNC_BUILTIN
  __builtin_amdgcn_global_load_async_to_lds_b128(
      (as1_v4i*)(uintptr_t)g, (as3_v4i*)(uint32_t)(uintptr_t)l, 0, 0);
#else
  asm volatile("global_load_async_to_lds_b128 %0, %1, off"
               :: "v"((uint32_t)(uintptr_t)l), "v"((uint64_t)(uintptr_t)g)
               : "memory");
#endif
}
__device__ __forceinline__ void async_wait0() {
#ifdef HAVE_WAITASYNC_BUILTIN
  __builtin_amdgcn_s_wait_asynccnt(0);
#else
  asm volatile("s_wait_asynccnt 0x0" ::: "memory");
#endif
}

// ---- VALU butterfly reductions within 16-lane half-groups (wave32) ----
// permlane16 nibble-select: lane i reads lane i^m (stays inside its row of 16),
// avoiding ds_bpermute traffic that competes with WMMA tile loads.
__device__ __forceinline__ float pl16_xor(float x, unsigned s0, unsigned s1) {
#ifdef HAVE_PERMLANE16
  int r = __builtin_amdgcn_permlane16(__float_as_int(x), __float_as_int(x),
                                      s0, s1, false, false);
  return __int_as_float(r);
#else
  (void)s0; (void)s1;
  return x;
#endif
}
__device__ __forceinline__ float hsum16(float x) {
#ifdef HAVE_PERMLANE16
  x += pl16_xor(x, 0x67452301u, 0xEFCDAB89u);  // ^1
  x += pl16_xor(x, 0x54761032u, 0xDCFE98BAu);  // ^2
  x += pl16_xor(x, 0x32107654u, 0xBA98FEDCu);  // ^4
  x += pl16_xor(x, 0xFEDCBA98u, 0x76543210u);  // ^8
#else
  x += __shfl_xor(x, 1, 32); x += __shfl_xor(x, 2, 32);
  x += __shfl_xor(x, 4, 32); x += __shfl_xor(x, 8, 32);
#endif
  return x;
}
__device__ __forceinline__ float hmax16(float x) {
#ifdef HAVE_PERMLANE16
  x = fmaxf(x, pl16_xor(x, 0x67452301u, 0xEFCDAB89u));
  x = fmaxf(x, pl16_xor(x, 0x54761032u, 0xDCFE98BAu));
  x = fmaxf(x, pl16_xor(x, 0x32107654u, 0xBA98FEDCu));
  x = fmaxf(x, pl16_xor(x, 0xFEDCBA98u, 0x76543210u));
#else
  x = fmaxf(x, __shfl_xor(x, 1, 32)); x = fmaxf(x, __shfl_xor(x, 2, 32));
  x = fmaxf(x, __shfl_xor(x, 4, 32)); x = fmaxf(x, __shfl_xor(x, 8, 32));
#endif
  return x;
}

// ---- WMMA fragment loaders (layouts per CDNA5 ISA 7.12.2, wave32) ----
__device__ __forceinline__ v16h fragA_lds(const _Float16* row, int hl) {
  const int kb = hl * 8;
  v8h lo = *(const v8h*)(row + kb);
  v8h hi = *(const v8h*)(row + kb + 16);
  return __builtin_shufflevector(lo, hi, 0,1,2,3,4,5,6,7,8,9,10,11,12,13,14,15);
}
__device__ __forceinline__ v16h fragB_lds(const _Float16* col, int hl) {
  const int kb = hl * 8;
  v8h lo = *(const v8h*)(col + kb);
  v8h hi = *(const v8h*)(col + kb + 16);
  return __builtin_shufflevector(lo, hi, 0,1,2,3,4,5,6,7,8,9,10,11,12,13,14,15);
}

// =====================================================================
// One-time fp32 -> fp16 elementwise convert (activations)
// =====================================================================
__global__ __launch_bounds__(256)
void cvt_f32_f16_k(const float* __restrict__ x, _Float16* __restrict__ y, int n4) {
  int i = blockIdx.x * 256 + threadIdx.x;
  if (i < n4) {
    float4 v = ((const float4*)x)[i];
    v4h o;
    o[0] = (_Float16)v.x; o[1] = (_Float16)v.y;
    o[2] = (_Float16)v.z; o[3] = (_Float16)v.w;
    ((v4h*)y)[i] = o;
  }
}

// =====================================================================
// One-time fp32 -> fp16 transposing convert: Wt[n][k] = (f16)W[k][n]
// =====================================================================
__global__ __launch_bounds__(256)
void transpose_f32_f16_k(const float* __restrict__ W, int ldw,
                         _Float16* __restrict__ Wt, int ldt, int K, int N) {
  __shared__ _Float16 t[32][33];
  const int k0 = blockIdx.y * 32, n0 = blockIdx.x * 32;
#pragma unroll
  for (int i = threadIdx.y; i < 32; i += 8)
    t[i][threadIdx.x] = (_Float16)W[(size_t)(k0 + i) * ldw + n0 + threadIdx.x];
  __syncthreads();
#pragma unroll
  for (int i = threadIdx.y; i < 32; i += 8)
    Wt[(size_t)(n0 + i) * ldt + k0 + threadIdx.x] = t[threadIdx.x][i];
}

// =====================================================================
// 128x128 tile GEMM, all-f16 operands:  C = A[M x K] * Bt[n][k]^T
// Double-buffered async global->LDS staging (DMA overlaps WMMA).
// Optional fused per-128-col RMS norm (each 128-col tile == one head).
// Block 256 thr = 8 waves (4x2), wave tile 32x64, k-step 64 (16 WMMA/step).
// =====================================================================
template<bool OUT16>
__global__ __launch_bounds__(256)
void gemm128_wmma_f16(const _Float16* __restrict__ A, int lda,
                      const _Float16* __restrict__ Bt, int ldb, int rowBase,
                      void* __restrict__ Cout, int ldc,
                      int M, int K,
                      const float* __restrict__ normW, float scale)
{
  __shared__ alignas(16) _Float16 As[2][128][72];   // ping-pong, 144B row stride
  __shared__ alignas(16) _Float16 Bs[2][128][72];
  __shared__ float rowss[128];

  const int tid  = threadIdx.x;
  const int wave = tid >> 5;
  const int lane = tid & 31;
  const int ln   = lane & 15;
  const int hl   = lane >> 4;
  const int wm   = wave >> 1;   // 0..3
  const int wn   = wave & 1;    // 0..1
  const int mBase = blockIdx.y * 128;
  const int nBase = blockIdx.x * 128;

  const v8f z8 = {0.f,0.f,0.f,0.f,0.f,0.f,0.f,0.f};
  v8f acc[2][4];
#pragma unroll
  for (int mt = 0; mt < 2; ++mt)
#pragma unroll
    for (int nt = 0; nt < 4; ++nt) acc[mt][nt] = z8;

  auto stage = [&](int k0, int buf) {
#pragma unroll
    for (int i = 0; i < 4; ++i) {
      int u  = i * 256 + tid;        // 1024 b128 units per tile
      int r  = u >> 3;               // row 0..127
      int sg = (u & 7) * 8;          // 8 halfs per unit
      int gr = mBase + r; if (gr >= M) gr = M - 1;   // clamp: rows store-guarded
      async_cp_b128(A  + (size_t)gr * lda + k0 + sg,                    &As[buf][r][sg]);
      async_cp_b128(Bt + (size_t)(rowBase + nBase + r) * ldb + k0 + sg, &Bs[buf][r][sg]);
    }
  };

  const int nk = K / 64;
  stage(0, 0);
  for (int kt = 0; kt < nk; ++kt) {
    async_wait0();          // this wave's copies for tile kt complete
    __syncthreads();        // everyone's copies complete; prev reads done
    if (kt + 1 < nk) stage((kt + 1) * 64, (kt + 1) & 1);  // DMA next tile
    const int buf = kt & 1;

    v16h af[2][2];
#pragma unroll
    for (int mt = 0; mt < 2; ++mt)
#pragma unroll
      for (int ks = 0; ks < 2; ++ks)
        af[mt][ks] = fragA_lds(&As[buf][wm*32 + mt*16 + ln][ks*32], hl);
#pragma unroll
    for (int nt = 0; nt < 4; ++nt) {
#pragma unroll
      for (int ks = 0; ks < 2; ++ks) {
        v16h bf = fragB_lds(&Bs[buf][wn*64 + nt*16 + ln][ks*32], hl);
#pragma unroll
        for (int mt = 0; mt < 2; ++mt)
          acc[mt][nt] = __builtin_amdgcn_wmma_f32_16x16x32_f16(
              false, af[mt][ks], false, bf, (short)0, acc[mt][nt], false, false);
      }
    }
  }
  __syncthreads();

  if (normW) {
    if (tid < 128) rowss[tid] = 0.f;
    __syncthreads();
#pragma unroll
    for (int mt = 0; mt < 2; ++mt) {
#pragma unroll
      for (int r = 0; r < 8; ++r) {
        float ss = 0.f;
#pragma unroll
        for (int nt = 0; nt < 4; ++nt) { float v = acc[mt][nt][r]; ss += v * v; }
        ss = hsum16(ss);
        if (ln == 0) atomicAdd(&rowss[wm*32 + mt*16 + hl*8 + r], ss);
      }
    }
    __syncthreads();
#pragma unroll
    for (int mt = 0; mt < 2; ++mt) {
#pragma unroll
      for (int r = 0; r < 8; ++r) {
        int row  = wm*32 + mt*16 + hl*8 + r;
        int grow = mBase + row;
        float inv = rsqrtf(rowss[row] * (1.0f/128.0f) + EPS_) * scale;
        if (grow < M) {
#pragma unroll
          for (int nt = 0; nt < 4; ++nt) {
            int d = wn*64 + nt*16 + ln;
            float v = acc[mt][nt][r] * inv * normW[d];
            size_t idx = (size_t)grow * ldc + nBase + d;
            if (OUT16) ((_Float16*)Cout)[idx] = (_Float16)v;
            else       ((float*)Cout)[idx]    = v;
          }
        }
      }
    }
  } else {
#pragma unroll
    for (int mt = 0; mt < 2; ++mt) {
#pragma unroll
      for (int r = 0; r < 8; ++r) {
        int row  = wm*32 + mt*16 + hl*8 + r;
        int grow = mBase + row;
        if (grow < M) {
#pragma unroll
          for (int nt = 0; nt < 4; ++nt) {
            int d = wn*64 + nt*16 + ln;
            float v = acc[mt][nt][r];
            size_t idx = (size_t)grow * ldc + nBase + d;
            if (OUT16) ((_Float16*)Cout)[idx] = (_Float16)v;
            else       ((float*)Cout)[idx]    = v;
          }
        }
      }
    }
  }
}

// =====================================================================
// Flash attention: grid (q_len/128, H). 8 waves; each wave owns 16 q-rows,
// fp32 acc 16x128. kv chunks of 64; K chunk async-copied (f16),
// V chunk transposed through LDS. SCALE pre-folded into Q.
// =====================================================================
__global__ __launch_bounds__(256)
void attn_wmma(const _Float16* __restrict__ Q, const _Float16* __restrict__ Kh,
               const _Float16* __restrict__ Vh, const float* __restrict__ mask,
               _Float16* __restrict__ O, int q_len, int kv_len)
{
  __shared__ alignas(16) _Float16 Kl[64][136];     // [kv][d]  B-frag for Q*K^T
  __shared__ alignas(16) _Float16 Vt[128][72];     // [d][kv]  B-frag for P*V
  __shared__ alignas(16) _Float16 Pl[8][16][72];   // per-wave P re-layout

  const int tid  = threadIdx.x;
  const int w    = tid >> 5;
  const int lane = tid & 31;
  const int ln   = lane & 15;
  const int hl   = lane >> 4;
  const int h    = blockIdx.y;
  const int kvh  = h >> 2;                 // h / G
  const int q0   = blockIdx.x * 128 + w * 16;

  // Q fragments straight from global f16 (A-frag pattern)
  v16h qf[4];
#pragma unroll
  for (int ds = 0; ds < 4; ++ds)
    qf[ds] = fragA_lds(Q + (size_t)(q0 + ln) * HD_ + h * D_ + ds * 32, hl);

  const v8f z8 = {0.f,0.f,0.f,0.f,0.f,0.f,0.f,0.f};
  const v8h z8h = {(_Float16)0,(_Float16)0,(_Float16)0,(_Float16)0,
                   (_Float16)0,(_Float16)0,(_Float16)0,(_Float16)0};
  v8f acc[8];
#pragma unroll
  for (int dt = 0; dt < 8; ++dt) acc[dt] = z8;
  float mrow[8], lrow[8];
#pragma unroll
  for (int r = 0; r < 8; ++r) { mrow[r] = -1e30f; lrow[r] = 0.f; }

  const int nch = (kv_len + 63) >> 6;
  for (int ck = 0; ck < nch; ++ck) {
    const int kvb = ck * 64;
    __syncthreads();
    // ---- stage K (async, layout-preserving) + V (transposed) ----
#pragma unroll
    for (int i = 0; i < 4; ++i) {
      int u  = i * 256 + tid;     // 1024 b128 units
      int r  = u >> 4;            // kv row 0..63
      int sg = (u & 15) * 8;      // d base 0..120
      int kv = kvb + r;
      if (kv < kv_len) {
        async_cp_b128(Kh + (size_t)kv * KVD_ + kvh * D_ + sg, &Kl[r][sg]);
        v8h vv = *(const v8h*)(Vh + (size_t)kv * KVD_ + kvh * D_ + sg);
#pragma unroll
        for (int j = 0; j < 8; ++j) Vt[sg + j][r] = vv[j];
      } else {
        *(v8h*)&Kl[r][sg] = z8h;
#pragma unroll
        for (int j = 0; j < 8; ++j) Vt[sg + j][r] = (_Float16)0.f;
      }
    }
    async_wait0();
    __syncthreads();

    // ---- S(16x64) = Q(16x128) * K^T ----
    v8f s[4];
#pragma unroll
    for (int nt = 0; nt < 4; ++nt) s[nt] = z8;
#pragma unroll
    for (int ds = 0; ds < 4; ++ds) {
#pragma unroll
      for (int nt = 0; nt < 4; ++nt) {
        v16h bf = fragB_lds(&Kl[nt*16 + ln][ds*32], hl);
        s[nt] = __builtin_amdgcn_wmma_f32_16x16x32_f16(
            false, qf[ds], false, bf, (short)0, s[nt], false, false);
      }
    }
    // ---- mask + kv bounds ----
#pragma unroll
    for (int nt = 0; nt < 4; ++nt) {
      int kv = kvb + nt*16 + ln;
#pragma unroll
      for (int r = 0; r < 8; ++r) {
        int q = q0 + r + 8*hl;
        float sv = s[nt][r];
        if (kv < kv_len) sv += mask[(size_t)q * kv_len + kv];
        else             sv  = -1e30f;
        s[nt][r] = sv;
      }
    }
    // ---- online softmax (VALU permlane16 reductions) ----
    float alpha[8];
#pragma unroll
    for (int r = 0; r < 8; ++r) {
      float mx = fmaxf(fmaxf(s[0][r], s[1][r]), fmaxf(s[2][r], s[3][r]));
      mx = hmax16(mx);
      mx = fmaxf(mx, mrow[r]);
      alpha[r] = __expf(mrow[r] - mx);
      mrow[r] = mx;
    }
#pragma unroll
    for (int nt = 0; nt < 4; ++nt)
#pragma unroll
      for (int r = 0; r < 8; ++r)
        s[nt][r] = __expf(s[nt][r] - mrow[r]);
#pragma unroll
    for (int r = 0; r < 8; ++r) {
      float ps = s[0][r] + s[1][r] + s[2][r] + s[3][r];
      ps = hsum16(ps);
      lrow[r] = lrow[r] * alpha[r] + ps;
#pragma unroll
      for (int dt = 0; dt < 8; ++dt) acc[dt][r] *= alpha[r];
    }
    // ---- P: D-frag -> A-frag via per-wave LDS ----
#pragma unroll
    for (int nt = 0; nt < 4; ++nt)
#pragma unroll
      for (int r = 0; r < 8; ++r)
        Pl[w][r + 8*hl][nt*16 + ln] = (_Float16)s[nt][r];

    v16h pa[2];
#pragma unroll
    for (int ks = 0; ks < 2; ++ks)
      pa[ks] = fragA_lds(&Pl[w][ln][ks*32], hl);
    // ---- acc += P(16x64) * V(64x128) ----
#pragma unroll
    for (int dt = 0; dt < 8; ++dt) {
#pragma unroll
      for (int ks = 0; ks < 2; ++ks) {
        v16h vb = fragB_lds(&Vt[dt*16 + ln][ks*32], hl);
        acc[dt] = __builtin_amdgcn_wmma_f32_16x16x32_f16(
            false, pa[ks], false, vb, (short)0, acc[dt], false, false);
      }
    }
  }

  // ---- epilogue ----
#pragma unroll
  for (int dt = 0; dt < 8; ++dt)
#pragma unroll
    for (int r = 0; r < 8; ++r) {
      int q = q0 + r + 8*hl;
      O[(size_t)q * HD_ + h * D_ + dt*16 + ln] = (_Float16)(acc[dt][r] / lrow[r]);
    }
}

// =====================================================================
extern "C" void kernel_launch(void* const* d_in, const int* in_sizes, int n_in,
                              void* d_out, int out_size, void* d_ws, size_t ws_size,
                              hipStream_t stream)
{
  const float* hidden = (const float*)d_in[0];
  const float* cross  = (const float*)d_in[1];
  const float* mask   = (const float*)d_in[2];
  const float* w_qkv  = (const float*)d_in[3];
  const float* w_o    = (const float*)d_in[4];
  const float* qnw    = (const float*)d_in[5];
  const float* knw    = (const float*)d_in[6];

  const int q_len  = in_sizes[0] / HID_;           // 1024
  const int kv_len = in_sizes[1] / HID_;           // 6404
  const int kvPad  = (kv_len + 127) & ~127;

  // fp16 workspace layout
  _Float16* Xh  = (_Float16*)d_ws;                           // q_len x HID
  _Float16* Xc  = Xh  + (size_t)q_len  * HID_;               // kv_len x HID
  _Float16* Wqt = Xc  + (size_t)kv_len * HID_;               // QKVN x HID (transposed)
  _Float16* Wot = Wqt + (size_t)QKVN_  * HID_;               // HID x HD  (transposed)
  _Float16* Qw  = Wot + (size_t)HID_   * HD_;                // q_len x HD
  _Float16* Kw  = Qw  + (size_t)q_len  * HD_;                // kvPad x KVD
  _Float16* Vw  = Kw  + (size_t)kvPad  * KVD_;               // kvPad x KVD
  _Float16* Ow  = Vw  + (size_t)kvPad  * KVD_;               // q_len x HD

  const int qmt  = q_len / 128;
  const int kvmt = (kv_len + 127) / 128;

  cvt_f32_f16_k<<<(in_sizes[0]/4 + 255)/256, 256, 0, stream>>>(hidden, Xh, in_sizes[0]/4);
  cvt_f32_f16_k<<<(in_sizes[1]/4 + 255)/256, 256, 0, stream>>>(cross,  Xc, in_sizes[1]/4);
  transpose_f32_f16_k<<<dim3(QKVN_/32, HID_/32), dim3(32,8), 0, stream>>>(
      w_qkv, QKVN_, Wqt, HID_, HID_, QKVN_);
  transpose_f32_f16_k<<<dim3(HID_/32, HD_/32), dim3(32,8), 0, stream>>>(
      w_o, HID_, Wot, HD_, HD_, HID_);

  // Q = RMSNorm(hidden @ Wq) * SCALE   (f16 out)
  gemm128_wmma_f16<true><<<dim3(HD_/128, qmt), 256, 0, stream>>>(
      Xh, HID_, Wqt, HID_, 0, Qw, HD_, q_len, HID_, qnw, SCALE_);
  // K = RMSNorm(cross @ Wk)            (f16 out)
  gemm128_wmma_f16<true><<<dim3(KVD_/128, kvmt), 256, 0, stream>>>(
      Xc, HID_, Wqt, HID_, HD_, Kw, KVD_, kv_len, HID_, knw, 1.0f);
  // V = cross @ Wv                     (f16 out)
  gemm128_wmma_f16<true><<<dim3(KVD_/128, kvmt), 256, 0, stream>>>(
      Xc, HID_, Wqt, HID_, HD_ + KVD_, Vw, KVD_, kv_len, HID_, nullptr, 1.0f);
  // Flash attention
  attn_wmma<<<dim3(qmt, H_), 256, 0, stream>>>(Qw, Kw, Vw, mask, Ow, q_len, kv_len);
  // out = O @ w_o                      (f32 out)
  gemm128_wmma_f16<false><<<dim3(HID_/128, qmt), 256, 0, stream>>>(
      Ow, HD_, Wot, HD_, 0, (float*)d_out, HID_, q_len, HD_, nullptr, 1.0f);
}
